// DMVFN_Simplified_67525475828099
// MI455X (gfx1250) — matrix operand order repaired and checked by reading, not compiled, not run on hardware
//
#include <hip/hip_runtime.h>

// ---------------------------------------------------------------------------
// DMVFN-simplified pipeline for MI455X (gfx1250, wave32, WMMA).
// Conv layers: implicit GEMM with v_wmma_f32_16x16x32_bf16.
//   A (16x32, M=pixels x K=Cin*9): im2col staged in LDS as bf16, K-major.
//     Pixel decode hoisted out of the K-loop (register descriptors);
//     K decode is once-per-thread-per-step since kk == lane for all rows.
//   B (32x16, K x N=Cout): weights pre-packed to bf16 [coPad][Kpad], loaded
//     directly from global (L2-resident) as 2x b128 per fragment + prefetch.
//   C/D (16x16 f32): 8 VGPRs/lane, fp32 accumulate; bias+relu (+optional
//     residual add) fused in the epilogue.
// Everything else (resize/warp/convT/gating) is plain VALU, memory-trivial.
// ---------------------------------------------------------------------------

#define B_  4
#define H_  256
#define W_  448
#define HW_ (H_ * W_)
#define PLANE ((long)B_ * HW_)   // 458752

typedef __attribute__((ext_vector_type(16))) __bf16          v16bf;
typedef __attribute__((ext_vector_type(8)))  float           v8f;
typedef __attribute__((ext_vector_type(8)))  unsigned short  us8;

union FragBf { us8 h[2]; v16bf v; };

__device__ __forceinline__ unsigned short f2bf(float f) {
    unsigned u = __float_as_uint(f);
    unsigned r = u + 0x7FFFu + ((u >> 16) & 1u);   // round-to-nearest-even
    return (unsigned short)(r >> 16);
}

__device__ __forceinline__ float clampf(float v, float lo, float hi) {
    return fminf(fmaxf(v, lo), hi);
}

// ---------------------------------------------------------------------------
// Generic utility kernels
// ---------------------------------------------------------------------------
__global__ void zero_k(float* p, long n) {
    long i = (long)blockIdx.x * blockDim.x + threadIdx.x;
    if (i < n) p[i] = 0.f;
}

// w0 = img0 (x ch 0..2), w1 = img1 (x ch 3..5)
__global__ void init_w01(const float* __restrict__ x, float* __restrict__ w0,
                         float* __restrict__ w1, long n) {   // n = B*3*HW
    long i = (long)blockIdx.x * blockDim.x + threadIdx.x;
    if (i >= n) return;
    long b = i / (3L * HW_);
    long r = i - b * 3L * HW_;
    w0[i] = x[b * 6L * HW_ + r];
    w1[i] = x[b * 6L * HW_ + 3L * HW_ + r];
}

// ---------------------------------------------------------------------------
// Gating head: relu(conv3x3(x6ch, wr)+br) spatial mean (batch 0 only),
// then 32->9 FC + sigmoid + normalize + round -> 9 gates.
// ---------------------------------------------------------------------------
__global__ void rconv_mean(const float* __restrict__ x, const float* __restrict__ wr,
                           const float* __restrict__ br, float* __restrict__ sums) {
    __shared__ float wsm[32 * 6 * 9];
    __shared__ float red[32];
    for (int i = threadIdx.x; i < 32 * 6 * 9; i += blockDim.x) wsm[i] = wr[i];
    if (threadIdx.x < 32) red[threadIdx.x] = 0.f;
    __syncthreads();
    int p = blockIdx.x * blockDim.x + threadIdx.x;   // pixel of batch 0
    if (p < HW_) {
        int y = p / W_, xx = p - y * W_;
        for (int c = 0; c < 32; ++c) {
            float a = br[c];
            for (int ci = 0; ci < 6; ++ci) {
                const float* sp = x + (long)ci * HW_;
                const float* wp = wsm + (c * 6 + ci) * 9;
                for (int ky = 0; ky < 3; ++ky) {
                    int iy = y + ky - 1;
                    if (iy < 0 || iy >= H_) continue;
                    for (int kx = 0; kx < 3; ++kx) {
                        int ix = xx + kx - 1;
                        if (ix < 0 || ix >= W_) continue;
                        a += sp[iy * W_ + ix] * wp[ky * 3 + kx];
                    }
                }
            }
            atomicAdd(&red[c], fmaxf(a, 0.f));
        }
    }
    __syncthreads();
    if (threadIdx.x < 32) atomicAdd(&sums[threadIdx.x], red[threadIdx.x]);
}

__global__ void gates_k(const float* __restrict__ sums, const float* __restrict__ wl,
                        const float* __restrict__ bl, float* __restrict__ gates) {
    if (threadIdx.x != 0 || blockIdx.x != 0) return;
    float r[32];
    for (int c = 0; c < 32; ++c) r[c] = sums[c] * (1.0f / (float)HW_);
    float v[9];
    float tot = 0.f;
    for (int j = 0; j < 9; ++j) {
        float a = bl[j];
        for (int c = 0; c < 32; ++c) a += r[c] * wl[j * 32 + c];
        v[j] = 1.f / (1.f + __expf(-a));
        tot += v[j];
    }
    for (int j = 0; j < 9; ++j) {
        float u = v[j] / (tot + 1e-6f) * 4.5f;
        gates[j] = rintf(clampf(u, 0.f, 1.f));
    }
}

// ---------------------------------------------------------------------------
// Pack 17-ch concat [img0,img1,w0,w1,mask_feed,flow_feed] and bilinear-resize
// from (H,W) to (sh,sw); flow channels (13..16) scaled by 1/scale.
// ---------------------------------------------------------------------------
__global__ void pack_resize(const float* __restrict__ x, const float* __restrict__ w0,
                            const float* __restrict__ w1, const float* __restrict__ mf,
                            const float* __restrict__ ff, float* __restrict__ xin,
                            int sh, int sw, float inv_s) {
    long idx = (long)blockIdx.x * blockDim.x + threadIdx.x;
    long shw = (long)sh * sw;
    long total = (long)B_ * 17 * shw;
    if (idx >= total) return;
    long pix = idx % shw;
    int c = (int)((idx / shw) % 17);
    int b = (int)(idx / (17 * shw));
    int oy = (int)(pix / sw), ox = (int)(pix - (long)oy * sw);

    float sy = clampf(((float)oy + 0.5f) * ((float)H_ / sh) - 0.5f, 0.f, (float)(H_ - 1));
    float sx = clampf(((float)ox + 0.5f) * ((float)W_ / sw) - 0.5f, 0.f, (float)(W_ - 1));
    int y0 = (int)floorf(sy), x0 = (int)floorf(sx);
    int y1 = min(y0 + 1, H_ - 1), x1 = min(x0 + 1, W_ - 1);
    float wy = sy - y0, wx = sx - x0;

    const float* src;
    float mult = 1.f;
    if (c < 6)        src = x  + ((long)b * 6 + c) * HW_;        // img0|img1
    else if (c < 9)   src = w0 + ((long)b * 3 + (c - 6)) * HW_;
    else if (c < 12)  src = w1 + ((long)b * 3 + (c - 9)) * HW_;
    else if (c == 12) src = mf + (long)b * HW_;
    else { src = ff + ((long)b * 4 + (c - 13)) * HW_; mult = inv_s; }

    float v = src[y0 * W_ + x0] * (1 - wx) * (1 - wy)
            + src[y0 * W_ + x1] * wx * (1 - wy)
            + src[y1 * W_ + x0] * (1 - wx) * wy
            + src[y1 * W_ + x1] * wx * wy;
    xin[idx] = v * mult;
}

// ---------------------------------------------------------------------------
// Weight prep: fp32 [Cout][K] (K = Cin*9 contiguous per co in OIHW) ->
// bf16 [coPad][Kpad], zero padded.
// ---------------------------------------------------------------------------
__global__ void wprep(const float* __restrict__ w, unsigned short* __restrict__ wbf,
                      int Cout, int K, int Kpad, int coPad) {
    int i = blockIdx.x * blockDim.x + threadIdx.x;
    if (i >= coPad * Kpad) return;
    int co = i / Kpad;
    int k  = i - co * Kpad;
    float v = (co < Cout && k < K) ? w[(long)co * K + k] : 0.f;
    wbf[i] = f2bf(v);
}

// ---------------------------------------------------------------------------
// Implicit-GEMM 3x3 conv (+bias,+relu,+optional residual) via WMMA bf16.
//   Block: 128 threads = 4 waves, M-tile 64 (wave w owns rows 16w..16w+15),
//   N = NT*16 column tiles, K stepped by 32.
// Fill mapping: element (t, thread) -> Alds row (wav + 4t), col `lane`; so
//   the K decode (ci,ky,kx) is computed once per thread per step, and the
//   pixel decode lives in 16 hoisted register descriptors.
// A layout (16-bit 16x32): lane<16 holds K 0..15 of row M=lane;
//   lane>=16 holds K 16..31 of row M=lane-16.
// B layout (16-bit 32x16): lane<16 holds K 0..15 of col N=lane;
//   lane>=16 holds K 16..31 of col N=lane-16.
// ---------------------------------------------------------------------------
template <int NT, int CIN, int STRIDE, int PAD>
__global__ __launch_bounds__(128)
void conv_wmma(const float* __restrict__ in, const unsigned short* __restrict__ wbf,
               const float* __restrict__ bias, const float* __restrict__ resid,
               float* __restrict__ out, int IH, int IW, int Cout, int OH, int OW) {
    constexpr int KTOT = CIN * 9;
    constexpr int KPAD = (KTOT + 31) & ~31;

    __shared__ __align__(16) unsigned short Alds[64 * 40];   // K-major, stride 40
    const int M     = B_ * OH * OW;
    const int mBase = blockIdx.x * 64;
    const int tid   = threadIdx.x;
    const int lane  = tid & 31;
    const int wav   = tid >> 5;
    const int half  = lane >> 4;
    const int l16   = lane & 15;
    const int ohow  = OH * OW;
    const int IHIW  = IH * IW;

    // ---- hoisted pixel descriptors: row (wav + 4t) of the A tile ----
    int fOff[16];    // b*CIN*IHIW + iy0*IW + ix0
    int fPack[16];   // (ix0 << 16) | (iy0 & 0xffff); invalid rows: iy0=-20000
    #pragma unroll
    for (int t = 0; t < 16; ++t) {
        int m  = mBase + wav + t * 4;
        int mm = (m < M) ? m : 0;
        int b  = mm / ohow;
        int r  = mm - b * ohow;
        int oy = r / OW;
        int ox = r - oy * OW;
        int iy0 = oy * STRIDE - PAD;
        int ix0 = ox * STRIDE - PAD;
        if (m >= M) iy0 = -20000;            // poisons bounds check below
        fOff[t]  = (b * CIN * IH + iy0) * IW + ix0;
        fPack[t] = (ix0 << 16) | (iy0 & 0xffff);
    }

    v8f acc[NT];
    v8f zero = {};
    #pragma unroll
    for (int nt = 0; nt < NT; ++nt) acc[nt] = zero;

    for (int k0 = 0; k0 < KPAD; k0 += 32) {
        __syncthreads();                     // previous iteration's frag reads done

        // ---- K decode: once per thread per step (kk == lane for all rows) ----
        int k  = k0 + lane;
        int ci = k / 9;                      // compile-time magic (k < KPAD const)
        int rem = k - ci * 9;
        int ky = rem / 3;
        int kx = rem - ky * 3;
        bool kvalid = (k < KTOT);
        int cstep = ci * IHIW + ky * IW + kx;

        // ---- im2col fill: 16 rows per thread, same column ----
        #pragma unroll
        for (int t = 0; t < 16; ++t) {
            int pack = fPack[t];
            int ix0 = pack >> 16;            // arithmetic shift
            int iy0 = (pack << 16) >> 16;    // sign-extended low half
            int iy = iy0 + ky, ix = ix0 + kx;
            float val = 0.f;
            if (kvalid && iy >= 0 && iy < IH && ix >= 0 && ix < IW)
                val = in[fOff[t] + cstep];
            Alds[(wav + t * 4) * 40 + lane] = f2bf(val);
        }
        __syncthreads();

        // ---- A fragment (this wave's 16 rows) ----
        FragBf fa;
        const unsigned short* arow = &Alds[(wav * 16 + l16) * 40];
        fa.h[0] = *(const us8*)(arow + half * 16);
        fa.h[1] = *(const us8*)(arow + half * 16 + 8);

        // ---- B fragments from packed bf16 weights (contiguous rows) ----
        #pragma unroll
        for (int nt = 0; nt < NT; ++nt) {
            const unsigned short* wrow =
                wbf + (long)(nt * 16 + l16) * KPAD + k0 + half * 16;
            if (k0 + 32 < KPAD) __builtin_prefetch(wrow + 32);
            FragBf fb;
            fb.h[0] = *(const us8*)wrow;
            fb.h[1] = *(const us8*)(wrow + 8);
            acc[nt] = __builtin_amdgcn_wmma_f32_16x16x32_bf16(
                false, fa.v, false, fb.v, (short)0, acc[nt], false, false);
        }
    }

    // ---- store: D lane holds col N = l16, rows M = half*8 + r ----
    #pragma unroll
    for (int nt = 0; nt < NT; ++nt) {
        int co = nt * 16 + l16;
        if (co >= Cout) continue;
        float bv = bias[co];
        #pragma unroll
        for (int r = 0; r < 8; ++r) {
            int m = mBase + wav * 16 + half * 8 + r;
            if (m >= M) continue;
            int b  = m / ohow;
            int rr = m - b * ohow;
            int oy = rr / OW;
            int ox = rr - oy * OW;
            long oidx = (((long)b * Cout + co) * OH + oy) * OW + ox;
            float v = fmaxf(acc[nt][r] + bv, 0.f);
            if (resid) v += resid[oidx];     // fused t = t + relu_chain
            out[oidx] = v;
        }
    }
}

// ---------------------------------------------------------------------------
// conv_transpose 4x4 stride 2 'SAME' (80 -> 5), gather form, pad_lo = 2.
// ---------------------------------------------------------------------------
__global__ void convT_k(const float* __restrict__ tin, const float* __restrict__ wlast,
                        const float* __restrict__ blast, float* __restrict__ outp,
                        int ih, int iw) {
    int oh = ih * 2, ow = iw * 2;
    long idx = (long)blockIdx.x * blockDim.x + threadIdx.x;
    long total = (long)B_ * 5 * oh * ow;
    if (idx >= total) return;
    long ohow = (long)oh * ow;
    long pix = idx % ohow;
    int co = (int)((idx / ohow) % 5);
    int b  = (int)(idx / (5 * ohow));
    int oy = (int)(pix / ow), ox = (int)(pix - (long)oy * ow);

    float acc = blast[co];
    long ihiw = (long)ih * iw;
    for (int ky = 0; ky < 4; ++ky) {
        int ty = oy + ky - 2;
        if (ty < 0 || (ty & 1) || (ty >> 1) >= ih) continue;
        int iy = ty >> 1;
        for (int kx = 0; kx < 4; ++kx) {
            int tx = ox + kx - 2;
            if (tx < 0 || (tx & 1) || (tx >> 1) >= iw) continue;
            int ix = tx >> 1;
            const float* tp = tin + (long)b * 80 * ihiw + (long)iy * iw + ix;
            const float* wp = wlast + (long)co * 1280 + ky * 4 + kx;
            float a = 0.f;
            for (int ci = 0; ci < 80; ++ci) a += tp[ci * ihiw] * wp[ci * 16];
            acc += a;
        }
    }
    outp[idx] = acc;
}

// ---------------------------------------------------------------------------
// Upsample tmp (B,5,ph,pw) to (H,W), gate, scale flow channels, accumulate
// into flow_sum/flow_feed/mask_sum/mask_feed.
// ---------------------------------------------------------------------------
__global__ void upacc_k(const float* __restrict__ tmp, const float* __restrict__ gate,
                        float fscale, float* __restrict__ flow_sum,
                        float* __restrict__ mask_sum, float* __restrict__ flow_feed,
                        float* __restrict__ mask_feed, int ph, int pw) {
    long idx = (long)blockIdx.x * blockDim.x + threadIdx.x;
    long total = (long)B_ * 5 * HW_;
    if (idx >= total) return;
    long pix = idx % HW_;
    int c = (int)((idx / HW_) % 5);
    int b = (int)(idx / (5L * HW_));
    int y = (int)(pix / W_), xx = (int)(pix - (long)y * W_);

    float sy = clampf(((float)y + 0.5f) * ((float)ph / H_) - 0.5f, 0.f, (float)(ph - 1));
    float sx = clampf(((float)xx + 0.5f) * ((float)pw / W_) - 0.5f, 0.f, (float)(pw - 1));
    int y0 = (int)floorf(sy), x0 = (int)floorf(sx);
    int y1 = min(y0 + 1, ph - 1), x1 = min(x0 + 1, pw - 1);
    float wy = sy - y0, wx = sx - x0;
    const float* src = tmp + ((long)b * 5 + c) * ph * pw;
    float v = src[y0 * pw + x0] * (1 - wx) * (1 - wy)
            + src[y0 * pw + x1] * wx * (1 - wy)
            + src[y1 * pw + x0] * (1 - wx) * wy
            + src[y1 * pw + x1] * wx * wy;
    float g = *gate;
    if (c < 4) {
        float fd = v * fscale * g;
        flow_sum[((long)b * 4 + c) * HW_ + pix]  += fd;
        flow_feed[((long)b * 4 + c) * HW_ + pix] += fd;
    } else {
        float md = v * g;
        mask_sum[(long)b * HW_ + pix]  += md;
        mask_feed[(long)b * HW_ + pix] += md;
    }
}

// ---------------------------------------------------------------------------
// Bilinear warp of a 3-ch image (from x, channel offset) with flow (2 planes).
// ---------------------------------------------------------------------------
__global__ void warp3_k(const float* __restrict__ x, int chOff,
                        const float* __restrict__ flow, int fc0,
                        float* __restrict__ dst) {
    long idx = (long)blockIdx.x * blockDim.x + threadIdx.x;
    if (idx >= (long)B_ * HW_) return;
    long pix = idx % HW_;
    int b = (int)(idx / HW_);
    int y = (int)(pix / W_), xx = (int)(pix - (long)y * W_);
    float fx = flow[((long)b * 4 + fc0) * HW_ + pix];
    float fy = flow[((long)b * 4 + fc0 + 1) * HW_ + pix];
    float sx = clampf((float)xx + fx, 0.f, (float)(W_ - 1));
    float sy = clampf((float)y + fy, 0.f, (float)(H_ - 1));
    int x0 = (int)floorf(sx), y0 = (int)floorf(sy);
    int x1 = min(x0 + 1, W_ - 1), y1 = min(y0 + 1, H_ - 1);
    float wx = sx - x0, wy = sy - y0;
    for (int c = 0; c < 3; ++c) {
        const float* s = x + ((long)b * 6 + chOff + c) * HW_;
        float v = s[y0 * W_ + x0] * (1 - wx) * (1 - wy)
                + s[y0 * W_ + x1] * wx * (1 - wy)
                + s[y1 * W_ + x0] * (1 - wx) * wy
                + s[y1 * W_ + x1] * wx * wy;
        dst[((long)b * 3 + c) * HW_ + pix] = v;
    }
}

// ---------------------------------------------------------------------------
// Final composite: clip(warp(img0,fs01)*sig(ms) + warp(img1,fs23)*(1-sig(ms))).
// ---------------------------------------------------------------------------
__global__ void final_k(const float* __restrict__ x, const float* __restrict__ fs,
                        const float* __restrict__ ms, float* __restrict__ out) {
    long idx = (long)blockIdx.x * blockDim.x + threadIdx.x;
    if (idx >= (long)B_ * HW_) return;
    long pix = idx % HW_;
    int b = (int)(idx / HW_);
    int y = (int)(pix / W_), xx = (int)(pix - (long)y * W_);
    float m = 1.f / (1.f + __expf(-ms[(long)b * HW_ + pix]));
    float fx0 = fs[((long)b * 4 + 0) * HW_ + pix], fy0 = fs[((long)b * 4 + 1) * HW_ + pix];
    float fx1 = fs[((long)b * 4 + 2) * HW_ + pix], fy1 = fs[((long)b * 4 + 3) * HW_ + pix];
    float sx0 = clampf((float)xx + fx0, 0.f, (float)(W_ - 1));
    float sy0 = clampf((float)y + fy0, 0.f, (float)(H_ - 1));
    float sx1 = clampf((float)xx + fx1, 0.f, (float)(W_ - 1));
    float sy1 = clampf((float)y + fy1, 0.f, (float)(H_ - 1));
    int ax0 = (int)floorf(sx0), ay0 = (int)floorf(sy0);
    int ax1 = min(ax0 + 1, W_ - 1), ay1 = min(ay0 + 1, H_ - 1);
    int bx0 = (int)floorf(sx1), by0 = (int)floorf(sy1);
    int bx1 = min(bx0 + 1, W_ - 1), by1 = min(by0 + 1, H_ - 1);
    float awx = sx0 - ax0, awy = sy0 - ay0;
    float bwx = sx1 - bx0, bwy = sy1 - by0;
    for (int c = 0; c < 3; ++c) {
        const float* s0 = x + ((long)b * 6 + c) * HW_;
        const float* s1 = x + ((long)b * 6 + 3 + c) * HW_;
        float v0 = s0[ay0 * W_ + ax0] * (1 - awx) * (1 - awy)
                 + s0[ay0 * W_ + ax1] * awx * (1 - awy)
                 + s0[ay1 * W_ + ax0] * (1 - awx) * awy
                 + s0[ay1 * W_ + ax1] * awx * awy;
        float v1 = s1[by0 * W_ + bx0] * (1 - bwx) * (1 - bwy)
                 + s1[by0 * W_ + bx1] * bwx * (1 - bwy)
                 + s1[by1 * W_ + bx0] * (1 - bwx) * bwy
                 + s1[by1 * W_ + bx1] * bwx * bwy;
        out[((long)b * 3 + c) * HW_ + pix] = clampf(v0 * m + v1 * (1.f - m), 0.f, 1.f);
    }
}

// ---------------------------------------------------------------------------
static inline int gsz(long n, int b) { return (int)((n + b - 1) / b); }

extern "C" void kernel_launch(void* const* d_in, const int* in_sizes, int n_in,
                              void* d_out, int out_size, void* d_ws, size_t ws_size,
                              hipStream_t stream) {
    (void)in_sizes; (void)n_in; (void)out_size; (void)ws_size;
    const float* x     = (const float*)d_in[0];
    const float* wr    = (const float*)d_in[1];
    const float* br    = (const float*)d_in[2];
    const float* wl    = (const float*)d_in[3];
    const float* bl    = (const float*)d_in[4];
    const float* w0a   = (const float*)d_in[5];
    const float* b0a   = (const float*)d_in[6];
    const float* w0b   = (const float*)d_in[7];
    const float* b0b   = (const float*)d_in[8];
    const float* wcb   = (const float*)d_in[9];
    const float* bcb   = (const float*)d_in[10];
    const float* wlast = (const float*)d_in[11];
    const float* blast = (const float*)d_in[12];
    float* out = (float*)d_out;

    float* ws = (float*)d_ws;
    float* w0buf = ws;                        // 3*PLANE
    float* w1buf = ws + 3 * PLANE;            // 3*PLANE
    float* maskf = ws + 6 * PLANE;            // 1*PLANE
    float* flowf = ws + 7 * PLANE;            // 4*PLANE
    float* flows = ws + 11 * PLANE;           // 4*PLANE
    float* masks = ws + 15 * PLANE;           // 1*PLANE
    float* xin   = ws + 16 * PLANE;           // 17*PLANE
    float* t0    = ws + 33 * PLANE;           // 10*PLANE (40ch @ 128x224)
    float* t1    = ws + 43 * PLANE;           // 5*PLANE  (80ch @ 64x112)
    float* r1    = ws + 48 * PLANE;           // 5*PLANE
    float* r2    = ws + 53 * PLANE;           // 5*PLANE
    float* tmp   = ws + 58 * PLANE;           // 2*PLANE (5ch @ 128x224)
    float* sums  = ws + 60 * PLANE;           // 32
    float* gates = ws + 60 * PLANE + 32;      // 9
    unsigned short* wbf = (unsigned short*)(ws + 60 * PLANE + 64); // 80*736 u16

    static const int scales[9] = {4, 4, 4, 2, 2, 2, 1, 1, 1};

    // ---- init feeds / sums / gating ----
    zero_k<<<gsz(10 * PLANE, 256), 256, 0, stream>>>(ws + 6 * PLANE, 10 * PLANE);
    zero_k<<<1, 64, 0, stream>>>(sums, 64);
    init_w01<<<gsz(3 * PLANE, 256), 256, 0, stream>>>(x, w0buf, w1buf, 3 * PLANE);
    rconv_mean<<<gsz(HW_, 256), 256, 0, stream>>>(x, wr, br, sums);
    gates_k<<<1, 1, 0, stream>>>(sums, wl, bl, gates);

    for (int i = 0; i < 9; ++i) {
        int s = scales[i];
        int sh = H_ / s, sw = W_ / s;
        int h2 = sh / 2, w2 = sw / 2;
        int h4 = sh / 4, w4 = sw / 4;

        // pack 17-ch input + resize (flow scaled by 1/s)
        long nxin = (long)B_ * 17 * sh * sw;
        pack_resize<<<gsz(nxin, 256), 256, 0, stream>>>(
            x, w0buf, w1buf, maskf, flowf, xin, sh, sw, 1.0f / (float)s);

        // conv0a: 17 -> 40 (pad 48), stride 2, pad_lo 0; K=153 pad 160
        wprep<<<gsz(48 * 160, 256), 256, 0, stream>>>(
            w0a + (long)i * 40 * 153, wbf, 40, 153, 160, 48);
        long M0 = (long)B_ * h2 * w2;
        conv_wmma<3, 17, 2, 0><<<gsz(M0, 64), 128, 0, stream>>>(
            xin, wbf, b0a + i * 40, nullptr, t0, sh, sw, 40, h2, w2);

        // conv0b: 40 -> 80, stride 2, pad_lo 0; K=360 pad 384
        wprep<<<gsz(80 * 384, 256), 256, 0, stream>>>(
            w0b + (long)i * 80 * 360, wbf, 80, 360, 384, 80);
        long M1 = (long)B_ * h4 * w4;
        conv_wmma<5, 40, 2, 0><<<gsz(M1, 64), 128, 0, stream>>>(
            t0, wbf, b0b + i * 80, nullptr, t1, h2, w2, 80, h4, w4);

        // residual chain: 4x (80 -> 80, stride 1, pad 1); K=720 pad 736.
        // j=3 writes t1 += relu(conv) (fused residual), so convT reads t1.
        const float* rin = t1;
        for (int j = 0; j < 4; ++j) {
            wprep<<<gsz(80 * 736, 256), 256, 0, stream>>>(
                wcb + ((long)i * 4 + j) * 80 * 720, wbf, 80, 720, 736, 80);
            float* rout = (j == 3) ? t1 : ((j & 1) ? r2 : r1);
            const float* resid = (j == 3) ? t1 : nullptr;
            conv_wmma<5, 80, 1, 1><<<gsz(M1, 64), 128, 0, stream>>>(
                rin, wbf, bcb + ((long)i * 4 + j) * 80, resid, rout,
                h4, w4, 80, h4, w4);
            rin = rout;
        }

        // conv_transpose 80 -> 5, x2 upsample
        convT_k<<<gsz((long)B_ * 5 * h2 * w2, 256), 256, 0, stream>>>(
            t1, wlast + (long)i * 6400, blast + i * 5, tmp, h4, w4);

        // upsample to (H,W), gate, accumulate flow/mask
        upacc_k<<<gsz((long)B_ * 5 * HW_, 256), 256, 0, stream>>>(
            tmp, gates + i, (float)(2 * s), flows, masks, flowf, maskf, h2, w2);

        // re-warp both images with updated flow_feed
        warp3_k<<<gsz((long)B_ * HW_, 256), 256, 0, stream>>>(x, 0, flowf, 0, w0buf);
        warp3_k<<<gsz((long)B_ * HW_, 256), 256, 0, stream>>>(x, 3, flowf, 2, w1buf);
    }

    final_k<<<gsz((long)B_ * HW_, 256), 256, 0, stream>>>(x, flows, masks, out);
}
